// MultiheadSelfAttention_70729521430565
// MI455X (gfx1250) — matrix operand order repaired and checked by reading, not compiled, not run on hardware
//
#include <hip/hip_runtime.h>

typedef _Float16 v16h __attribute__((ext_vector_type(16)));
typedef _Float16 v8h  __attribute__((ext_vector_type(8)));
typedef float    v8f  __attribute__((ext_vector_type(8)));
typedef unsigned v4u  __attribute__((ext_vector_type(4)));
typedef int      v8i  __attribute__((ext_vector_type(8)));
typedef int      v4i  __attribute__((ext_vector_type(4)));

union AFrag { v16h v; v8h h[2]; _Float16 e[16]; };

#define D_MODEL 1024
#define NHEAD   16
#define HEAD_DIM 64

// ---- CDNA5 async global->LDS copy of 16 bytes (per lane), GV addressing ----
__device__ __forceinline__ void async_copy16(void* lds, const void* g) {
  const unsigned loff = (unsigned)(size_t)lds;   // generic LDS ptr low 32b = LDS offset
  asm volatile("global_load_async_to_lds_b128 %0, %1, off"
               :: "v"(loff), "v"((unsigned long long)(size_t)g)
               : "memory");
}
__device__ __forceinline__ void wait_async0() {
  asm volatile("s_wait_asynccnt 0x0" ::: "memory");
}

// ---- CDNA5 Tensor Data Mover: 2D tile (row-major) global -> LDS ------------
// tile_d1 rows of tile_d0 elements (2B), row stride = stride_d0 elements.
#if __has_builtin(__builtin_amdgcn_tensor_load_to_lds)
#define HAVE_TDM 1
__device__ __forceinline__ void tdm_load_2d(void* lds, const void* gaddr,
                                            unsigned tile_d0, unsigned tile_d1,
                                            unsigned stride_d0) {
  const unsigned long long ga = (unsigned long long)(size_t)gaddr;
  v4u g0;
  g0[0] = 1u;                                   // count=1 (valid), user mode
  g0[1] = (unsigned)(size_t)lds;                // lds_addr (byte offset)
  g0[2] = (unsigned)ga;                         // global_addr[31:0]
  g0[3] = (unsigned)(ga >> 32) | (2u << 30);    // global_addr[56:32] | type=2
  v8i g1;
  g1[0] = (int)(1u << 16);                      // data_size=1 (2 bytes/elem)
  g1[1] = (int)0xFFFF0000u;                     // tensor_dim0 = 0x7FFFFFFF (lo16)
  g1[2] = (int)0xFFFF7FFFu;                     // tensor_dim0 hi | tensor_dim1 lo
  g1[3] = (int)(0x7FFFu | (tile_d0 << 16));     // tensor_dim1 hi | tile_dim0
  g1[4] = (int)tile_d1;                         // tile_dim1 | tile_dim2=0
  g1[5] = (int)stride_d0;                       // tensor_dim0_stride[31:0]
  g1[6] = 0;
  g1[7] = 0;
  __builtin_amdgcn_tensor_load_to_lds(g0, g1, (v4i)0, (v4i)0, (v8i)0, 0);
}
#endif

// ---- 16-lane (half-wave) max reduction via DPP butterflies (no LDS) --------
__device__ __forceinline__ float rowmax16(float x) {
  int v;
  v = __builtin_amdgcn_update_dpp(0, __builtin_bit_cast(int, x), 0xB1, 0xF, 0xF, true);  // xor 1
  x = fmaxf(x, __builtin_bit_cast(float, v));
  v = __builtin_amdgcn_update_dpp(0, __builtin_bit_cast(int, x), 0x4E, 0xF, 0xF, true);  // xor 2
  x = fmaxf(x, __builtin_bit_cast(float, v));
  v = __builtin_amdgcn_update_dpp(0, __builtin_bit_cast(int, x), 0x141, 0xF, 0xF, true); // half mirror
  x = fmaxf(x, __builtin_bit_cast(float, v));
  v = __builtin_amdgcn_update_dpp(0, __builtin_bit_cast(int, x), 0x140, 0xF, 0xF, true); // mirror
  x = fmaxf(x, __builtin_bit_cast(float, v));
  return x;
}

// ---------------------------------------------------------------- f32 -> f16
__global__ void cvt_f32_to_f16(const float* __restrict__ src,
                               _Float16* __restrict__ dst, int n) {
  int i = blockIdx.x * blockDim.x + threadIdx.x;
  if (i < n) dst[i] = (_Float16)src[i];
}

// ---------------------------------------------------------------- LayerNorm
__global__ __launch_bounds__(256)
void ln_kernel(const float* __restrict__ x, const float* __restrict__ gamma,
               const float* __restrict__ beta, _Float16* __restrict__ xn) {
  const int D = D_MODEL;
  const int row = blockIdx.x;
  const float* xr = x + (size_t)row * D;
  float v[4]; float s = 0.f, ss = 0.f;
#pragma unroll
  for (int i = 0; i < 4; ++i) {
    v[i] = xr[threadIdx.x + i * 256];
    s += v[i]; ss += v[i] * v[i];
  }
#pragma unroll
  for (int off = 16; off > 0; off >>= 1) {
    s  += __shfl_xor(s,  off, 32);
    ss += __shfl_xor(ss, off, 32);
  }
  __shared__ float rs[8], rss[8];
  const int w = threadIdx.x >> 5, lane = threadIdx.x & 31;
  if (lane == 0) { rs[w] = s; rss[w] = ss; }
  __syncthreads();
  if (threadIdx.x == 0) {
    float a = 0.f, b2 = 0.f;
    for (int i = 0; i < 8; ++i) { a += rs[i]; b2 += rss[i]; }
    rs[0] = a; rss[0] = b2;
  }
  __syncthreads();
  const float mean = rs[0] * (1.f / 1024.f);
  const float var  = rss[0] * (1.f / 1024.f) - mean * mean;
  const float rstd = rsqrtf(var + 1e-5f);
#pragma unroll
  for (int i = 0; i < 4; ++i) {
    const int c = threadIdx.x + i * 256;
    xn[(size_t)row * D + c] = (_Float16)(((v[i] - mean) * rstd) * gamma[c] + beta[c]);
  }
}

// ------------------------------------------------- generic WMMA GEMM + bias
// C[M x N] = A[M x K](f16,row) @ B[K x N](f16,row) + bias
// 128x64 block tile, 256 threads = 8 waves (4m x 2n), each wave 32x32.
// Double-buffered LDS: async-DMA A tile + coalesced/scattered B tile for
// iteration k+1 issued before computing iteration k.
template <bool OUT_HALF>
__global__ __launch_bounds__(256)
void gemm128x64(const _Float16* __restrict__ A, const _Float16* __restrict__ Bm,
                const float* __restrict__ bias, void* __restrict__ Cout,
                int M, int N, int K) {
  __shared__ alignas(32) _Float16 As[2][128][32];
  __shared__ alignas(32) _Float16 Bt[2][64][32];   // column-major: Bt[buf][n][k]
  const int tid  = threadIdx.x;
  const int wave = tid >> 5, lane = tid & 31;
  const int lrow = lane & 15, lhalf = lane >> 4;
  const int m0 = blockIdx.x * 128, n0 = blockIdx.y * 64;
  const int wm = (wave >> 1) * 32, wn = (wave & 1) * 32;

  const int ar = tid >> 1, ac = (tid & 1) * 16;   // A staging: 128 rows x 2 chunks
  const int bk = tid >> 3, bn8 = (tid & 7) * 8;   // B staging: 32 k-rows x 8 cols

  // prologue: stage tile 0
  async_copy16(&As[0][ar][ac], A + (size_t)(m0 + ar) * K + ac);
  {
    const v8h bv = *(const v8h*)(Bm + (size_t)bk * N + (n0 + bn8));
#pragma unroll
    for (int i = 0; i < 8; ++i) Bt[0][bn8 + i][bk] = bv[i];
  }
  wait_async0();
  __syncthreads();

  v8f acc[2][2] = {};
  for (int k0 = 0; k0 < K; k0 += 32) {
    const int cur = (k0 >> 5) & 1, nxt = cur ^ 1;
    const bool has_next = (k0 + 32) < K;
    v8h bv;
    if (has_next) {   // issue next tile's traffic before computing
      async_copy16(&As[nxt][ar][ac], A + (size_t)(m0 + ar) * K + (k0 + 32 + ac));
      bv = *(const v8h*)(Bm + (size_t)(k0 + 32 + bk) * N + (n0 + bn8));
    }

#pragma unroll
    for (int mi = 0; mi < 2; ++mi) {
      AFrag a;
      const _Float16* ap = &As[cur][wm + mi * 16 + lrow][lhalf * 8];
      a.h[0] = *(const v8h*)ap;
      a.h[1] = *(const v8h*)(ap + 16);
#pragma unroll
      for (int ni = 0; ni < 2; ++ni) {
        AFrag b;
        const _Float16* bp = &Bt[cur][wn + ni * 16 + lrow][lhalf * 8];
        b.h[0] = *(const v8h*)bp;
        b.h[1] = *(const v8h*)(bp + 16);
        acc[mi][ni] = __builtin_amdgcn_wmma_f32_16x16x32_f16(
            false, a.v, false, b.v, (short)0, acc[mi][ni], false, false);
      }
    }

    if (has_next) {
#pragma unroll
      for (int i = 0; i < 8; ++i) Bt[nxt][bn8 + i][bk] = bv[i];
    }
    wait_async0();
    __syncthreads();
  }

#pragma unroll
  for (int mi = 0; mi < 2; ++mi)
#pragma unroll
    for (int ni = 0; ni < 2; ++ni) {
      const int col  = n0 + wn + ni * 16 + lrow;
      const float bc = bias[col];
#pragma unroll
      for (int r = 0; r < 8; ++r) {
        const int row   = m0 + wm + mi * 16 + lhalf * 8 + r;
        const float val = acc[mi][ni][r] + bc;
        if (OUT_HALF) ((_Float16*)Cout)[(size_t)row * N + col] = (_Float16)val;
        else          ((float*)Cout)[(size_t)row * N + col]    = val;
      }
    }
}

// ------------------------------------------- fused flash attention (f16 WMMA)
// qkv: [B*T, 3072] f16; ctx: [B*T, 1024] f16.  8 waves/WG, wave = 16 q-rows.
// K tile moved by the Tensor Data Mover (one descriptor per 32-key block),
// V transposed via coalesced read + DS scatter, both double-buffered.
// V is extended with a 5th all-ones N-tile so the P@V WMMA accumulates the
// softmax denominator in o[4] (no cross-lane sum reductions).
__global__ __launch_bounds__(256)
void attn_fused(const _Float16* __restrict__ qkv, _Float16* __restrict__ ctx, int T) {
  const int D3 = 3 * D_MODEL;
  const int b = blockIdx.x >> 4, h = blockIdx.x & 15;
  const int qbase = blockIdx.y * 128;
  const int tid = threadIdx.x, wave = tid >> 5, lane = tid & 31;
  const int lrow = lane & 15, lhalf = lane >> 4;

  __shared__ alignas(32) _Float16 Ks[2][32][64];   // [buf][key][d] row-major
  __shared__ alignas(32) _Float16 Vt[2][80][32];   // [buf][d][key]; rows 64..79 = 1
  __shared__ alignas(32) _Float16 Ps[8][16][32];   // per-wave P transpose scratch

  { // ones tiles (never overwritten)
    const int r = 64 + (tid >> 4), c = (tid & 15) * 2;
    Vt[0][r][c] = (_Float16)1.f; Vt[0][r][c + 1] = (_Float16)1.f;
    Vt[1][r][c] = (_Float16)1.f; Vt[1][r][c + 1] = (_Float16)1.f;
  }

  const _Float16* baseq = qkv + (size_t)(b * T) * D3 + h * HEAD_DIM;

  // Q fragments (A-layout), 1/sqrt(dh)=0.125 folded in (exact in f16)
  AFrag qf[2];
  {
    const _Float16* qp = baseq + (size_t)(qbase + wave * 16 + lrow) * D3;
#pragma unroll
    for (int c = 0; c < 2; ++c) {
      const int d0 = c * 32 + lhalf * 8;
      qf[c].h[0] = *(const v8h*)(qp + d0);
      qf[c].h[1] = *(const v8h*)(qp + d0 + 16);
#pragma unroll
      for (int i = 0; i < 16; ++i) qf[c].e[i] = qf[c].e[i] * (_Float16)0.125f;
    }
  }

  float m_run[8];
#pragma unroll
  for (int r = 0; r < 8; ++r) m_run[r] = -1e30f;
  v8f o[5] = {};   // o[0..3]=ctx accum, o[4]=softmax denominator accum

  const int key_ld = tid >> 3, dc8 = (tid & 7) * 8;

  // prologue: stage block 0
#if HAVE_TDM
  if (wave == 0) tdm_load_2d(&Ks[0][0][0], baseq + D_MODEL, 64, 32, (unsigned)D3);
#else
  async_copy16(&Ks[0][key_ld][dc8], baseq + D_MODEL + (size_t)key_ld * D3 + dc8);
#endif
  {
    const v8h vv = *(const v8h*)(baseq + 2 * D_MODEL + (size_t)key_ld * D3 + dc8);
#pragma unroll
    for (int i = 0; i < 8; ++i) Vt[0][dc8 + i][key_ld] = vv[i];
  }
#if HAVE_TDM
  __builtin_amdgcn_s_wait_tensorcnt(0);
#else
  wait_async0();
#endif
  __syncthreads();

  for (int kb = 0; kb < T; kb += 32) {
    const int cur = (kb >> 5) & 1, nxt = cur ^ 1;
    const bool has_next = (kb + 32) < T;
    v8h vv;
    if (has_next) {   // issue next block's K (TDM) + V read before computing
#if HAVE_TDM
      if (wave == 0)
        tdm_load_2d(&Ks[nxt][0][0], baseq + D_MODEL + (size_t)(kb + 32) * D3,
                    64, 32, (unsigned)D3);
#else
      async_copy16(&Ks[nxt][key_ld][dc8],
                   baseq + D_MODEL + (size_t)(kb + 32 + key_ld) * D3 + dc8);
#endif
      vv = *(const v8h*)(baseq + 2 * D_MODEL + (size_t)(kb + 32 + key_ld) * D3 + dc8);
    }

    // S(16x32) = Qs(16x64) @ K^T(64x32): 2 k-chunks x 2 n-tiles
    v8f sa[2] = {};
#pragma unroll
    for (int c = 0; c < 2; ++c)
#pragma unroll
      for (int ni = 0; ni < 2; ++ni) {
        AFrag kf;
        const _Float16* kp = &Ks[cur][ni * 16 + lrow][c * 32 + lhalf * 8];
        kf.h[0] = *(const v8h*)kp;
        kf.h[1] = *(const v8h*)(kp + 16);
        sa[ni] = __builtin_amdgcn_wmma_f32_16x16x32_f16(
            false, qf[c].v, false, kf.v, (short)0, sa[ni], false, false);
      }

    // online softmax: DPP row-max, exp, rescale; denominator rides in o[4]
#pragma unroll
    for (int r = 0; r < 8; ++r) {
      const float s0 = sa[0][r];
      const float s1 = sa[1][r];
      const float mx = rowmax16(fmaxf(s0, s1));
      const float m_new = fmaxf(m_run[r], mx);
      const float corr  = __expf(m_run[r] - m_new);
      m_run[r] = m_new;
#pragma unroll
      for (int ni = 0; ni < 5; ++ni) o[ni][r] *= corr;
      Ps[wave][lhalf * 8 + r][lrow]      = (_Float16)__expf(s0 - m_new);
      Ps[wave][lhalf * 8 + r][16 + lrow] = (_Float16)__expf(s1 - m_new);
    }
    // wave-private LDS round trip: DS is in-order within a wave, only need a
    // compiler ordering barrier (no workgroup barrier).
    asm volatile("" ::: "memory");

    // O(16x64)|L(16x16) += P(16x32) @ [V | 1](32x80): K=32, 5 n-tiles
    AFrag pf;
    {
      const _Float16* pp = &Ps[wave][lrow][lhalf * 8];
      pf.h[0] = *(const v8h*)pp;
      pf.h[1] = *(const v8h*)(pp + 16);
    }
#pragma unroll
    for (int ni = 0; ni < 5; ++ni) {
      AFrag vf;
      const _Float16* vp2 = &Vt[cur][ni * 16 + lrow][lhalf * 8];
      vf.h[0] = *(const v8h*)vp2;
      vf.h[1] = *(const v8h*)(vp2 + 16);
      o[ni] = __builtin_amdgcn_wmma_f32_16x16x32_f16(
          false, pf.v, false, vf.v, (short)0, o[ni], false, false);
    }

    if (has_next) {
#pragma unroll
      for (int i = 0; i < 8; ++i) Vt[nxt][dc8 + i][key_ld] = vv[i];
    }
#if HAVE_TDM
    __builtin_amdgcn_s_wait_tensorcnt(0);
#else
    wait_async0();
#endif
    __syncthreads();
  }

  // epilogue: normalize by accumulated denominator, scatter into ctx
#pragma unroll
  for (int r = 0; r < 8; ++r) {
    const float inv = 1.f / fmaxf(o[4][r], 1e-12f);
    const int q = qbase + wave * 16 + lhalf * 8 + r;
    _Float16* op = ctx + (size_t)(b * T + q) * D_MODEL + h * HEAD_DIM;
#pragma unroll
    for (int ni = 0; ni < 4; ++ni)
      op[ni * 16 + lrow] = (_Float16)(o[ni][r] * inv);
  }
}

// ---------------------------------------------------------------- launcher
extern "C" void kernel_launch(void* const* d_in, const int* in_sizes, int n_in,
                              void* d_out, int out_size, void* d_ws, size_t ws_size,
                              hipStream_t stream) {
  const float* x        = (const float*)d_in[0];
  const float* ln_gamma = (const float*)d_in[1];
  const float* ln_beta  = (const float*)d_in[2];
  const float* W_qkv    = (const float*)d_in[3];
  const float* b_qkv    = (const float*)d_in[4];
  const float* W_proj   = (const float*)d_in[5];
  const float* b_proj   = (const float*)d_in[6];
  float* out = (float*)d_out;

  const int D  = D_MODEL;
  const int BT = in_sizes[0] / D;   // 4096
  const int Bb = 2;
  const int T  = BT / Bb;           // 2048

  char* ws = (char*)d_ws;
  size_t off = 0;
  auto walloc = [&](size_t bytes) -> void* {
    void* p = ws + off;
    off += (bytes + 255) & ~(size_t)255;
    return p;
  };
  _Float16* xn_h    = (_Float16*)walloc((size_t)BT * D * 2);        //  8 MB
  _Float16* wqkv_h  = (_Float16*)walloc((size_t)D * 3 * D * 2);     //  6 MB
  _Float16* wproj_h = (_Float16*)walloc((size_t)D * D * 2);         //  2 MB
  _Float16* qkv_h   = (_Float16*)walloc((size_t)BT * 3 * D * 2);    // 24 MB
  _Float16* ctx_h   = (_Float16*)walloc((size_t)BT * D * 2);        //  8 MB

  { const int n = D * 3 * D;
    cvt_f32_to_f16<<<(n + 255) / 256, 256, 0, stream>>>(W_qkv, wqkv_h, n); }
  { const int n = D * D;
    cvt_f32_to_f16<<<(n + 255) / 256, 256, 0, stream>>>(W_proj, wproj_h, n); }

  ln_kernel<<<BT, 256, 0, stream>>>(x, ln_gamma, ln_beta, xn_h);

  gemm128x64<true><<<dim3(BT / 128, (3 * D) / 64), 256, 0, stream>>>(
      xn_h, wqkv_h, b_qkv, qkv_h, BT, 3 * D, D);

  attn_fused<<<dim3(Bb * NHEAD, T / 128), 256, 0, stream>>>(qkv_h, ctx_h, T);

  gemm128x64<false><<<dim3(BT / 128, D / 64), 256, 0, stream>>>(
      ctx_h, wproj_h, b_proj, out, BT, D, D);
}
